// VRNN_7413113553332
// MI455X (gfx1250) — matrix operand order repaired and checked by reading, not compiled, not run on hardware
//
#include <hip/hip_runtime.h>
#include <hip/hip_bf16.h>

// ---------------- problem constants ----------------
constexpr int T_STEPS = 150;
constexpr int BATCH   = 4096;
constexpr int XDIM    = 13;
constexpr int HDIM    = 128;
constexpr int ZDIM    = 32;
constexpr int BTILE   = 32;    // batch rows per workgroup
constexpr int NWAVE   = 8;     // 256 threads = 8 wave32
constexpr float EPSV   = 1e-3f;
constexpr float CLAMPV = 1e-9f;

// ---------------- packed weight offsets (halves), layout [Opad x Ipad] row-major
constexpr int W_PHI_X1  = 0;       // 128 x 32
constexpr int W_PHI_X2  = 4096;    // 128 x 128
constexpr int W_PHI_Z   = 20480;   // 128 x 32
constexpr int W_ENC1    = 24576;   // 128 x 256
constexpr int W_ENC2    = 57344;   // 128 x 128
constexpr int W_ENC_MU  = 73728;   // 32 x 128
constexpr int W_ENC_STD = 77824;   // 32 x 128
constexpr int W_PRIOR   = 81920;   // 128 x 128
constexpr int W_PR_MU   = 98304;   // 32 x 128
constexpr int W_PR_STD  = 102400;  // 32 x 128
constexpr int W_DEC1    = 106496;  // 128 x 256
constexpr int W_DEC2    = 139264;  // 128 x 128
constexpr int W_DM1     = 155648;  // 16 x 128
constexpr int W_DM2     = 157696;  // 16 x 32
constexpr int W_DSTD    = 158208;  // 16 x 128 (unused by losses, packed anyway)
constexpr int W_GRU_IH  = 160256;  // 384 x 256
constexpr int W_GRU_HH  = 258560;  // 384 x 128
constexpr int W_TOTAL_HALVES = 307712;
constexpr size_t BIAS_BYTE_OFF = (size_t)W_TOTAL_HALVES * 2;  // 615424, 16B aligned

// bias blob offsets (floats)
constexpr int B_PHI_X1 = 0,   B_PHI_X2 = 128, B_PHI_Z = 256, B_ENC1 = 384,
              B_ENC2   = 512, B_ENC_MU = 640, B_ENC_STD = 672, B_PRIOR = 704,
              B_PR_MU  = 832, B_PR_STD = 864, B_DEC1 = 896, B_DEC2 = 1024,
              B_DM1    = 1152, B_DM2 = 1168, B_DSTD = 1184, BIAS_TOTAL = 1200;

// ---------------- vector types ----------------
typedef __attribute__((ext_vector_type(16))) _Float16 v16h;
typedef __attribute__((ext_vector_type(8)))  _Float16 v8h;
typedef __attribute__((ext_vector_type(8)))  float    v8f;

__device__ __forceinline__ v8f wmma_f16(v16h a, v16h b, v8f c) {
  return __builtin_amdgcn_wmma_f32_16x16x32_f16(false, a, false, b, (short)0, c,
                                                false, false);
}

// A fragment (16x32 f16, M x K) from a row-major [rows x ldk] f16 LDS tile.
// Layout: lanes 0-15: M=lane, K 0..15 consecutive;
// lanes 16-31: same M, K 16..31.
__device__ __forceinline__ v16h load_a(const _Float16* A, int ldk, int m0,
                                       int k0, int lane) {
  const _Float16* p = A + (size_t)(m0 + (lane & 15)) * ldk + k0 + ((lane >> 4) << 4);
  v8h lo = *(const v8h*)(p);        // ds_load_b128
  v8h hi = *(const v8h*)(p + 8);    // ds_load_b128
  return __builtin_shufflevector(lo, hi, 0,1,2,3,4,5,6,7,8,9,10,11,12,13,14,15);
}

// B fragment (32x16 f16, K x N) for C=A@W^T: B[k,n]=W[n,k].
// Layout: lanes 0-15 hold N=lane, K 0..15 consecutive; lanes 16-31: K 16..31.
// With W packed [Opad x Ipad] row-major, each lane reads 16 contiguous halves.
__device__ __forceinline__ v16h load_b(const _Float16* W, int ipad, int n0,
                                       int k0, int lane) {
  const _Float16* p = W + (size_t)(n0 + (lane & 15)) * ipad + k0 + ((lane >> 4) << 4);
  return *(const v16h*)(p);         // 2x global_load_b128 (32B aligned)
}

// C/D layout: lane holds N = lane&15; VGPR g holds M = m0 + g + 8*(lane>=16).
// ACT: 0 = relu -> f16, 1 = linear -> f32, 2 = softplus -> f32
template <int ACT>
__device__ __forceinline__ void store_c(void* out, int ldo, int m0, int n0,
                                        v8f c, const float* bias, int lane) {
  int n = n0 + (lane & 15);
  float bv = bias[n];
  int mb = m0 + ((lane >> 4) << 3);
#pragma unroll
  for (int g = 0; g < 8; ++g) {
    float v = c[g] + bv;
    if (ACT == 0) {
      v = v > 0.f ? v : 0.f;
      ((_Float16*)out)[(size_t)(mb + g) * ldo + n] = (_Float16)v;
    } else if (ACT == 1) {
      ((float*)out)[(size_t)(mb + g) * ldo + n] = v;
    } else {
      float s = (v > 20.f) ? v : log1pf(__expf(v));
      ((float*)out)[(size_t)(mb + g) * ldo + n] = s;
    }
  }
}

// GEMM over a 32-row batch tile: out[32 x NT*16] = act(A[32 x KT*32] @ W^T + b)
// `wave` must be wave-uniform (SGPR) so the tile loop is scalar and EXEC stays
// all-ones around the WMMAs. All K-step fragments are preloaded so the LDS +
// global loads issue back-to-back before the first wait.
template <int KT, int NT, int ACT>
__device__ __forceinline__ void gemm(const _Float16* A, int ldk,
                                     const _Float16* W, int ipad,
                                     const float* bias, void* out, int ldo,
                                     int wave, int lane) {
  for (int tile = wave; tile < 2 * NT; tile += NWAVE) {
    int mt = tile & 1, nt = tile >> 1;
    v16h af[KT], bf[KT];
#pragma unroll
    for (int kt = 0; kt < KT; ++kt) {
      af[kt] = load_a(A, ldk, mt * 16, kt * 32, lane);
      bf[kt] = load_b(W, ipad, nt * 16, kt * 32, lane);
    }
    v8f c = {};
#pragma unroll
    for (int kt = 0; kt < KT; ++kt) c = wmma_f16(af[kt], bf[kt], c);
    store_c<ACT>(out, ldo, mt * 16, nt * 16, c, bias, lane);
  }
}

// GEMM with concatenated A = [A1 | A2] (avoids a cat buffer).
template <int KT1, int KT2, int NT, int ACT>
__device__ __forceinline__ void gemm_cat(const _Float16* A1, int ld1,
                                         const _Float16* A2, int ld2,
                                         const _Float16* W, int ipad,
                                         const float* bias, void* out, int ldo,
                                         int wave, int lane) {
  for (int tile = wave; tile < 2 * NT; tile += NWAVE) {
    int mt = tile & 1, nt = tile >> 1;
    v8f c = {};
    {
      v16h af[KT1], bf[KT1];
#pragma unroll
      for (int kt = 0; kt < KT1; ++kt) {
        af[kt] = load_a(A1, ld1, mt * 16, kt * 32, lane);
        bf[kt] = load_b(W, ipad, nt * 16, kt * 32, lane);
      }
#pragma unroll
      for (int kt = 0; kt < KT1; ++kt) c = wmma_f16(af[kt], bf[kt], c);
    }
    {
      v16h af[KT2], bf[KT2];
#pragma unroll
      for (int kt = 0; kt < KT2; ++kt) {
        af[kt] = load_a(A2, ld2, mt * 16, kt * 32, lane);
        bf[kt] = load_b(W, ipad, nt * 16, (KT1 + kt) * 32, lane);
      }
#pragma unroll
      for (int kt = 0; kt < KT2; ++kt) c = wmma_f16(af[kt], bf[kt], c);
    }
    store_c<ACT>(out, ldo, mt * 16, nt * 16, c, bias, lane);
  }
}

__device__ __forceinline__ float sigf(float x) { return 1.f / (1.f + __expf(-x)); }

// ---------------- weight packing ----------------
struct PackArgs {
  const float* W[17];
  const float* Bb[15];
};

__device__ static const int PK_O[17]  = {128,128,128,128,128,32,32,128,32,32,128,128,13,13,13,384,384};
__device__ static const int PK_I[17]  = {13,128,32,256,128,128,128,128,128,128,256,128,128,13,128,256,128};
__device__ static const int PK_OP[17] = {128,128,128,128,128,32,32,128,32,32,128,128,16,16,16,384,384};
__device__ static const int PK_IP[17] = {32,128,32,256,128,128,128,128,128,128,256,128,128,32,128,256,128};
__device__ static const int PK_OFF[17] = {W_PHI_X1,W_PHI_X2,W_PHI_Z,W_ENC1,W_ENC2,
  W_ENC_MU,W_ENC_STD,W_PRIOR,W_PR_MU,W_PR_STD,W_DEC1,W_DEC2,W_DM1,W_DM2,W_DSTD,
  W_GRU_IH,W_GRU_HH};
__device__ static const int PK_BO[15]  = {128,128,128,128,128,32,32,128,32,32,128,128,13,13,13};
__device__ static const int PK_BOP[15] = {128,128,128,128,128,32,32,128,32,32,128,128,16,16,16};
__device__ static const int PK_BOFF[15] = {B_PHI_X1,B_PHI_X2,B_PHI_Z,B_ENC1,B_ENC2,
  B_ENC_MU,B_ENC_STD,B_PRIOR,B_PR_MU,B_PR_STD,B_DEC1,B_DEC2,B_DM1,B_DM2,B_DSTD};

__global__ void vrnn_pack(PackArgs pa, _Float16* wdst, float* bdst) {
  int gstride = gridDim.x * blockDim.x;
  int gtid = blockIdx.x * blockDim.x + threadIdx.x;
  for (int l = 0; l < 17; ++l) {
    int n = PK_OP[l] * PK_IP[l];
    for (int idx = gtid; idx < n; idx += gstride) {
      int o = idx / PK_IP[l], i = idx - o * PK_IP[l];
      float v = (o < PK_O[l] && i < PK_I[l]) ? pa.W[l][(size_t)o * PK_I[l] + i] : 0.f;
      wdst[PK_OFF[l] + idx] = (_Float16)v;
    }
  }
  for (int l = 0; l < 15; ++l) {
    for (int idx = gtid; idx < PK_BOP[l]; idx += gstride) {
      bdst[PK_BOFF[l] + idx] = (idx < PK_BO[l]) ? pa.Bb[l][idx] : 0.f;
    }
  }
}

__global__ void vrnn_zero(float* out_total) {
  if (threadIdx.x == 0 && blockIdx.x == 0) *out_total = 0.f;
}

// ---------------- persistent VRNN kernel ----------------
__global__ __launch_bounds__(256, 1) void vrnn_main(
    const float* __restrict__ x, const float* __restrict__ noise,
    const _Float16* __restrict__ Wp, const float* __restrict__ Bp,
    float* __restrict__ out_total, float* __restrict__ out_mu) {
  const int batch0 = blockIdx.x * BTILE;
  const int tid  = threadIdx.x;
  // wave id is identical across the 32 lanes of a wave: pin it to an SGPR so
  // tile loops are scalar and EXEC stays all-ones around WMMA.
  const int wave = __builtin_amdgcn_readfirstlane(tid >> 5);
  const int lane = tid & 31;

  // ---- LDS arena (~58 KB) ----
  __shared__ float    s_bias[BIAS_TOTAL];
  __shared__ _Float16 s_x[BTILE * 32];          // x_t padded 13->32
  __shared__ _Float16 s_h[BTILE * HDIM];        // hidden state (f16)
  __shared__ _Float16 s_bufB[BTILE * HDIM];     // phi_x1out / enc1out / pr_h / dec1out
  __shared__ _Float16 s_phix[BTILE * HDIM];     // phi_x (lives whole step)
  __shared__ _Float16 s_bufD[BTILE * HDIM];     // enc_h, then phi_z
  __shared__ __align__(16) char s_bufE[BTILE * HDIM * 2];  // enc_mu/std f32 -> dec_h f16
  __shared__ __align__(16) char s_bufF[BTILE * HDIM * 2];  // pr_mu/std f32 -> dm1 f16 + dec_mu f32
  __shared__ _Float16 s_z[BTILE * ZDIM];
  __shared__ float    s_red[256];

  float* emu  = (float*)s_bufE;           // [32 x 32]
  float* estd = emu + BTILE * ZDIM;       // [32 x 32]
  float* pmu  = (float*)s_bufF;
  float* pstd = pmu + BTILE * ZDIM;
  _Float16* dech  = (_Float16*)s_bufE;    // [32 x 128]
  _Float16* dm1   = (_Float16*)s_bufF;    // [32 x 32] (cols 13..31 zero)
  float*    dmu32 = (float*)(s_bufF + 4096);  // [32 x 16]

  for (int i = tid; i < BIAS_TOTAL; i += 256) s_bias[i] = Bp[i];
  for (int i = tid; i < BTILE * HDIM; i += 256) s_h[i] = (_Float16)0.f;
  __syncthreads();

  float acc_kld = 0.f, acc_nll = 0.f, acc_rec = 0.f;

  for (int t = 0; t < T_STEPS; ++t) {
    // keep GRU weights warm in cache for this step
    __builtin_prefetch(Wp + W_GRU_IH, 0, 1);
    __builtin_prefetch(Wp + W_GRU_HH, 0, 1);

    // ---- load x tile (pad 13 -> 32), f16 ----
    const float* xt = x + ((size_t)t * BATCH + batch0) * XDIM;
    for (int i = tid; i < BTILE * 32; i += 256) {
      int m = i >> 5, j = i & 31;
      float v = (j < XDIM) ? xt[m * XDIM + j] : 0.f;
      s_x[i] = (_Float16)v;
    }
    __syncthreads();

    // phi_x = relu(phi_x2 @ relu(phi_x1 @ x))
    gemm<1, 8, 0>(s_x, 32, Wp + W_PHI_X1, 32, s_bias + B_PHI_X1, s_bufB, HDIM, wave, lane);
    __syncthreads();
    gemm<4, 8, 0>(s_bufB, HDIM, Wp + W_PHI_X2, HDIM, s_bias + B_PHI_X2, s_phix, HDIM, wave, lane);
    __syncthreads();

    // enc_h = relu(enc2 @ relu(enc1 @ [phi_x, h]))
    gemm_cat<4, 4, 8, 0>(s_phix, HDIM, s_h, HDIM, Wp + W_ENC1, 256, s_bias + B_ENC1,
                         s_bufB, HDIM, wave, lane);
    __syncthreads();
    gemm<4, 8, 0>(s_bufB, HDIM, Wp + W_ENC2, HDIM, s_bias + B_ENC2, s_bufD, HDIM, wave, lane);
    __syncthreads();

    // enc_mu / enc_std (softplus) ; pr_h = relu(prior @ h)
    gemm<4, 2, 1>(s_bufD, HDIM, Wp + W_ENC_MU, HDIM, s_bias + B_ENC_MU, emu, ZDIM, wave, lane);
    gemm<4, 2, 2>(s_bufD, HDIM, Wp + W_ENC_STD, HDIM, s_bias + B_ENC_STD, estd, ZDIM, wave, lane);
    gemm<4, 8, 0>(s_h, HDIM, Wp + W_PRIOR, HDIM, s_bias + B_PRIOR, s_bufB, HDIM, wave, lane);
    __syncthreads();
    gemm<4, 2, 1>(s_bufB, HDIM, Wp + W_PR_MU, HDIM, s_bias + B_PR_MU, pmu, ZDIM, wave, lane);
    gemm<4, 2, 2>(s_bufB, HDIM, Wp + W_PR_STD, HDIM, s_bias + B_PR_STD, pstd, ZDIM, wave, lane);
    __syncthreads();

    // ---- z sample + KLD (elementwise) ----
    const float* nt_ptr = noise + ((size_t)t * BATCH + batch0) * ZDIM;
    for (int i = tid; i < BTILE * ZDIM; i += 256) {
      int m = i >> 5, j = i & 31;
      float mu1 = emu[i], lv = estd[i], mu2 = pmu[i], ps = pstd[i];
      float zv = mu1 + nt_ptr[m * ZDIM + j] * __expf(0.5f * lv);
      s_z[i] = (_Float16)zv;
      float s1 = fmaxf(lv, CLAMPV), s2 = fmaxf(ps, CLAMPV);
      float d = mu1 - mu2;
      acc_kld += 0.5f * (2.f * __logf(s2 + EPSV) - 2.f * __logf(s1 + EPSV) +
                         (s1 * s1 + d * d) / (s2 * s2) - 1.f);
    }
    __syncthreads();

    // phi_z = relu(phi_z @ z)
    gemm<1, 8, 0>(s_z, 32, Wp + W_PHI_Z, 32, s_bias + B_PHI_Z, s_bufD, HDIM, wave, lane);
    __syncthreads();

    // dec_h = relu(dec2 @ relu(dec1 @ [phi_z, h]))
    gemm_cat<4, 4, 8, 0>(s_bufD, HDIM, s_h, HDIM, Wp + W_DEC1, 256, s_bias + B_DEC1,
                         s_bufB, HDIM, wave, lane);
    __syncthreads();
    gemm<4, 8, 0>(s_bufB, HDIM, Wp + W_DEC2, HDIM, s_bias + B_DEC2, dech, HDIM, wave, lane);
    __syncthreads();

    // dec_mu = dec_mean2 @ relu(dec_mean1 @ dec_h)   (dec_std unused by losses)
    for (int i = tid; i < BTILE * 16; i += 256)       // zero pad cols 16..31
      dm1[(i >> 4) * 32 + 16 + (i & 15)] = (_Float16)0.f;
    gemm<4, 1, 0>(dech, HDIM, Wp + W_DM1, HDIM, s_bias + B_DM1, dm1, 32, wave, lane);
    __syncthreads();
    gemm<1, 1, 1>(dm1, 32, Wp + W_DM2, 32, s_bias + B_DM2, dmu32, 16, wave, lane);
    __syncthreads();

    // ---- dec_mus output + NLL + rec (elementwise) ----
    for (int i = tid; i < BTILE * XDIM; i += 256) {
      int m = i / XDIM, j = i - m * XDIM;
      float mu = dmu32[m * 16 + j];
      out_mu[((size_t)t * BATCH + batch0 + m) * XDIM + j] = mu;
      float xv = (float)s_x[m * 32 + j];
      float dmx = mu - xv;
      acc_rec += dmx * dmx;
      float th = fmaxf(mu, CLAMPV);
      acc_nll -= xv * __logf(th + EPSV) + (1.f - xv) * __logf(1.f - th + EPSV);
    }

    // ---- fused GRU: 6 gate tiles per output tile, all in registers ----
    float hnew[2][8];
#pragma unroll
    for (int it = 0; it < 2; ++it) {
      int tile = wave + it * NWAVE;
      int mt = tile & 1, nt = tile >> 1;
      v8f cir = {}, ciz = {}, cin = {}, chr_ = {}, chz = {}, chn = {};
      // phase 1: gi from phi_x (K columns 0..127 of Wih)
      {
        v16h af[4];
#pragma unroll
        for (int kt = 0; kt < 4; ++kt)
          af[kt] = load_a(s_phix, HDIM, mt * 16, kt * 32, lane);
#pragma unroll
        for (int kt = 0; kt < 4; ++kt) {
          cir = wmma_f16(af[kt], load_b(Wp + W_GRU_IH, 256, nt * 16, kt * 32, lane), cir);
          ciz = wmma_f16(af[kt], load_b(Wp + W_GRU_IH, 256, HDIM + nt * 16, kt * 32, lane), ciz);
          cin = wmma_f16(af[kt], load_b(Wp + W_GRU_IH, 256, 2 * HDIM + nt * 16, kt * 32, lane), cin);
        }
      }
      // phase 2: gi from phi_z (K columns 128..255 of Wih)
      {
        v16h af[4];
#pragma unroll
        for (int kt = 0; kt < 4; ++kt)
          af[kt] = load_a(s_bufD, HDIM, mt * 16, kt * 32, lane);
#pragma unroll
        for (int kt = 0; kt < 4; ++kt) {
          cir = wmma_f16(af[kt], load_b(Wp + W_GRU_IH, 256, nt * 16, 128 + kt * 32, lane), cir);
          ciz = wmma_f16(af[kt], load_b(Wp + W_GRU_IH, 256, HDIM + nt * 16, 128 + kt * 32, lane), ciz);
          cin = wmma_f16(af[kt], load_b(Wp + W_GRU_IH, 256, 2 * HDIM + nt * 16, 128 + kt * 32, lane), cin);
        }
      }
      // phase 3: gh from h
      {
        v16h af[4];
#pragma unroll
        for (int kt = 0; kt < 4; ++kt)
          af[kt] = load_a(s_h, HDIM, mt * 16, kt * 32, lane);
#pragma unroll
        for (int kt = 0; kt < 4; ++kt) {
          chr_ = wmma_f16(af[kt], load_b(Wp + W_GRU_HH, HDIM, nt * 16, kt * 32, lane), chr_);
          chz  = wmma_f16(af[kt], load_b(Wp + W_GRU_HH, HDIM, HDIM + nt * 16, kt * 32, lane), chz);
          chn  = wmma_f16(af[kt], load_b(Wp + W_GRU_HH, HDIM, 2 * HDIM + nt * 16, kt * 32, lane), chn);
        }
      }
      int n = nt * 16 + (lane & 15);
      int mb = mt * 16 + ((lane >> 4) << 3);
#pragma unroll
      for (int g = 0; g < 8; ++g) {
        float r  = sigf(cir[g] + chr_[g]);
        float zg = sigf(ciz[g] + chz[g]);
        float nn = tanhf(cin[g] + r * chn[g]);
        float ho = (float)s_h[(size_t)(mb + g) * HDIM + n];
        hnew[it][g] = (1.f - zg) * nn + zg * ho;
      }
    }
    __syncthreads();  // all reads of h done before overwrite
#pragma unroll
    for (int it = 0; it < 2; ++it) {
      int tile = wave + it * NWAVE;
      int mt = tile & 1, nt = tile >> 1;
      int n = nt * 16 + (lane & 15);
      int mb = mt * 16 + ((lane >> 4) << 3);
#pragma unroll
      for (int g = 0; g < 8; ++g)
        s_h[(size_t)(mb + g) * HDIM + n] = (_Float16)hnew[it][g];
    }
    __syncthreads();
  }

  // ---- loss reduction: rec is a global mean over T*B*XD ----
  s_red[tid] = acc_rec * (1.f / (float)(T_STEPS * BATCH * XDIM)) + acc_kld + acc_nll;
  __syncthreads();
  for (int s = 128; s > 0; s >>= 1) {
    if (tid < s) s_red[tid] += s_red[tid + s];
    __syncthreads();
  }
  if (tid == 0) atomicAdd(out_total, s_red[0]);
}

// ---------------- launcher ----------------
extern "C" void kernel_launch(void* const* d_in, const int* in_sizes, int n_in,
                              void* d_out, int out_size, void* d_ws, size_t ws_size,
                              hipStream_t stream) {
  (void)in_sizes; (void)n_in; (void)out_size; (void)ws_size;
  const float* x     = (const float*)d_in[0];
  const float* noise = (const float*)d_in[1];

  PackArgs pa;
  for (int l = 0; l < 15; ++l) {             // 15 (W,b) layers in dict order
    pa.W[l]  = (const float*)d_in[2 + 2 * l];
    pa.Bb[l] = (const float*)d_in[3 + 2 * l];
  }
  pa.W[15] = (const float*)d_in[32];          // gru_ih (no bias)
  pa.W[16] = (const float*)d_in[33];          // gru_hh (no bias)

  _Float16* wpack = (_Float16*)d_ws;
  float*    bpack = (float*)((char*)d_ws + BIAS_BYTE_OFF);
  float*    out   = (float*)d_out;

  vrnn_zero<<<1, 32, 0, stream>>>(out);
  vrnn_pack<<<64, 256, 0, stream>>>(pa, wpack, bpack);
  vrnn_main<<<BATCH / BTILE, 256, 0, stream>>>(x, noise, wpack, bpack, out, out + 1);
}
